// RNN_3848290697661
// MI455X (gfx1250) — matrix operand order repaired and checked by reading, not compiled, not run on hardware
//
#include <hip/hip_runtime.h>
#include <cmath>

typedef __bf16 bf16_t;
typedef __attribute__((ext_vector_type(16))) __bf16 bf16x16;
typedef __attribute__((ext_vector_type(8)))  __bf16 bf16x8;
typedef __attribute__((ext_vector_type(8)))  float  v8f;
typedef __attribute__((ext_vector_type(4)))  float  v4f;

namespace {
constexpr int T_STEPS = 2048;
constexpr int NB = 128;        // batch
constexpr int ND = 128;        // input dim
constexpr int NH = 256;        // hidden dim
constexpr int HPAD = NH + 8;   // 264 bf16 = 132 dwords/row -> conflict-free-ish strides
constexpr int DPAD = ND + 8;   // 136 bf16 = 68 dwords/row
}

// Branch-free tanh: tanh(x) = 1 - 2/(exp(2x)+1), exp via v_exp_f32 (2^u).
// TRANS32 ops co-execute with the XDL WMMA pipe; no EXEC divergence.
__device__ __forceinline__ float fast_tanh(float x) {
  float e2 = __builtin_amdgcn_exp2f(x * 2.88539008177792681f);  // 2*log2(e)
  return 1.0f - 2.0f * __builtin_amdgcn_rcpf(e2 + 1.0f);
}

__launch_bounds__(512, 1)
__global__ void rnn_fused_scan(const float* __restrict__ x,
                               const float* __restrict__ h0,
                               const float* __restrict__ W_in,
                               const float* __restrict__ b_in,
                               const float* __restrict__ W_h,
                               const float* __restrict__ b_h,
                               float* __restrict__ out)
{
  // double-buffered hidden state and staged x tile (bf16, A-matrix friendly row-major)
  __shared__ __align__(16) bf16_t hbuf[2][16][HPAD];
  __shared__ __align__(16) bf16_t xbuf[2][16][DPAD];

  const int tid  = (int)threadIdx.x;
  const int wave = tid >> 5;          // 0..15 -> N-tile index
  const int lane = tid & 31;
  const int half = lane >> 4;         // 0 or 1
  const int lr   = lane & 15;
  const int n0   = wave * 16;
  const int b0   = (int)blockIdx.x * 16;

  // ---------------- persistent weight fragments (B-matrix layout) ----------------
  // B 32x16 bf16: lane l holds column n0 + (l%16); element e <-> K = kc*32 + (l/16)*16 + e
  bf16x16 wh[8];
#pragma unroll
  for (int kc = 0; kc < 8; ++kc) {
    const float* p = W_h + (size_t)(kc * 32 + half * 16) * NH + (n0 + lr);
    bf16x16 f;
#pragma unroll
    for (int e = 0; e < 16; ++e) f[e] = (bf16_t)p[(size_t)e * NH];
    wh[kc] = f;
  }
  bf16x16 wi[4];
#pragma unroll
  for (int kc = 0; kc < 4; ++kc) {
    const float* p = W_in + (size_t)(kc * 32 + half * 16) * NH + (n0 + lr);
    bf16x16 f;
#pragma unroll
    for (int e = 0; e < 16; ++e) f[e] = (bf16_t)p[(size_t)e * NH];
    wi[kc] = f;
  }
  const float bias = b_in[n0 + lr] + b_h[n0 + lr];  // per accumulator column

  // ---------------- init h state and stage x[0] ----------------
  {
    const float* hp = h0 + (size_t)(b0 + wave) * NH + lane * 8;
    v4f ha = *(const v4f*)hp;
    v4f hb = *(const v4f*)(hp + 4);
#pragma unroll
    for (int j = 0; j < 4; ++j) {
      hbuf[0][wave][lane * 8 + j]     = (bf16_t)ha[j];
      hbuf[0][wave][lane * 8 + 4 + j] = (bf16_t)hb[j];
    }
    const float* xp = x + (size_t)(b0 + wave) * ND + lane * 4;
    v4f x0 = *(const v4f*)xp;
#pragma unroll
    for (int j = 0; j < 4; ++j) xbuf[0][wave][lane * 4 + j] = (bf16_t)x0[j];
  }
  __syncthreads();

  v4f xn = {};
  for (int t = 0; t < T_STEPS; ++t) {
    const int cur = t & 1;
    const int nxt = cur ^ 1;

    // software pipeline: fetch x[t+1] while computing step t
    if (t + 1 < T_STEPS) {
      const float* xp = x + ((size_t)(t + 1) * NB + b0 + wave) * ND + lane * 4;
      xn = *(const v4f*)xp;
    }
    if (t + 2 < T_STEPS) {
      __builtin_prefetch(x + ((size_t)(t + 2) * NB + b0 + wave) * ND + lane * 4, 0, 1);
    }

    // three accumulators to shorten the dependent WMMA chain (depth 4 instead of 12)
    v8f acc0, acc1, acc2;
#pragma unroll
    for (int i = 0; i < 8; ++i) { acc0[i] = bias; acc1[i] = 0.f; acc2[i] = 0.f; }

    // x[t] @ W_in  : 4 k-chunks of 32
#pragma unroll
    for (int kc = 0; kc < 4; ++kc) {
      const bf16_t* pr = &xbuf[cur][lr][kc * 32 + half * 16];
      bf16x8 lo = *(const bf16x8*)pr;          // K = kc*32 + half*16 + [0..7]
      bf16x8 hi = *(const bf16x8*)(pr + 8);    // K = kc*32 + half*16 + 8 + [0..7]
      bf16x16 a;
#pragma unroll
      for (int e = 0; e < 8; ++e) { a[e] = lo[e]; a[e + 8] = hi[e]; }
      acc0 = __builtin_amdgcn_wmma_f32_16x16x32_bf16(false, a, false, wi[kc],
                                                     (short)0, acc0, false, false);
    }
    // h[t] @ W_h   : 8 k-chunks of 32
#pragma unroll
    for (int kc = 0; kc < 8; ++kc) {
      const bf16_t* pr = &hbuf[cur][lr][kc * 32 + half * 16];
      bf16x8 lo = *(const bf16x8*)pr;
      bf16x8 hi = *(const bf16x8*)(pr + 8);
      bf16x16 a;
#pragma unroll
      for (int e = 0; e < 8; ++e) { a[e] = lo[e]; a[e + 8] = hi[e]; }
      if (kc < 4)
        acc1 = __builtin_amdgcn_wmma_f32_16x16x32_bf16(false, a, false, wh[kc],
                                                       (short)0, acc1, false, false);
      else
        acc2 = __builtin_amdgcn_wmma_f32_16x16x32_bf16(false, a, false, wh[kc],
                                                       (short)0, acc2, false, false);
    }

    // tanh, emit h_seq[t] (fp32) and next h state (bf16 in LDS)
    // C layout: VGPR i -> row (half*8 + i), col (n0 + lr)
    float* orow = out + (size_t)t * NB * NH + (size_t)(b0 + half * 8) * NH + (n0 + lr);
#pragma unroll
    for (int i = 0; i < 8; ++i) {
      float v = fast_tanh(acc0[i] + acc1[i] + acc2[i]);
      orow[(size_t)i * NH] = v;
      hbuf[nxt][half * 8 + i][n0 + lr] = (bf16_t)v;
    }

    // commit staged x[t+1] to LDS (bf16)
    if (t + 1 < T_STEPS) {
#pragma unroll
      for (int j = 0; j < 4; ++j) xbuf[nxt][wave][lane * 4 + j] = (bf16_t)xn[j];
    }
    __syncthreads();   // single barrier per timestep (double-buffered LDS)
  }
}

extern "C" void kernel_launch(void* const* d_in, const int* in_sizes, int n_in,
                              void* d_out, int out_size, void* d_ws, size_t ws_size,
                              hipStream_t stream) {
  const float* x    = (const float*)d_in[0];
  const float* h0   = (const float*)d_in[1];
  const float* W_in = (const float*)d_in[2];
  const float* b_in = (const float*)d_in[3];
  const float* W_h  = (const float*)d_in[4];
  const float* b_h  = (const float*)d_in[5];
  float* out = (float*)d_out;

  dim3 grid(NB / 16);   // 8 workgroups, one per 16 batch rows (independent recurrences)
  dim3 block(512);      // 16 wave32s, one per 16-wide slice of H
  hipLaunchKernelGGL(rnn_fused_scan, grid, block, 0, stream,
                     x, h0, W_in, b_in, W_h, b_h, out);
}
